// MultilevelCrossAttentionBlockWithRoPE_33878702031453
// MI455X (gfx1250) — matrix operand order repaired and checked by reading, not compiled, not run on hardware
//
#include <hip/hip_runtime.h>

typedef unsigned short u16;
typedef unsigned int   u32;

#define B_  4
#define LQ_ 512
#define D_  256
#define H_  8
#define DH_ 32
#define S_  5440   // 64*64+32*32+16*16+8*8  (170 chunks of 32 keys)

typedef __attribute__((ext_vector_type(16))) __bf16 v16bf;
typedef __attribute__((ext_vector_type(8)))  float  v8f;

union Frag { uint4 u[2]; v16bf v; };

__device__ __forceinline__ u16 f2bf(float f) {
    u32 u = __float_as_uint(f);
    u32 r = (u + 0x7FFFu + ((u >> 16) & 1u)) >> 16;   // round-to-nearest-even
    return (u16)r;
}

__device__ __forceinline__ v8f wmma_bf16(v16bf a, v16bf b, v8f c) {
    // 8 args: (neg_a, A, neg_b, B, c_mod, C, reuse_a, reuse_b)
    return __builtin_amdgcn_wmma_f32_16x16x32_bf16(false, a, false, b, (short)0, c, false, false);
}

// A fragment (16x32 bf16 tile): p -> &tile[row(lane&15) * stride + chunk*32]
// lanes 0-15: K = 0..15 ; lanes 16-31: K = 16..31
__device__ __forceinline__ v16bf a_frag(const u16* p, int lane) {
    int kb = (lane & 16) ? 16 : 0;
    Frag f;
    f.u[0] = *reinterpret_cast<const uint4*>(p + kb);
    f.u[1] = *reinterpret_cast<const uint4*>(p + kb + 8);
    return f.v;
}

// B fragment (32x16 bf16, KxN): per-lane 16 contiguous K values starting at p
__device__ __forceinline__ v16bf b_frag(const u16* p) {
    Frag f;
    f.u[0] = *reinterpret_cast<const uint4*>(p);
    f.u[1] = *reinterpret_cast<const uint4*>(p + 8);
    return f.v;
}

// --------------------------------------------------- all four weights -> bf16
__global__ __launch_bounds__(256)
void k_cvt4(const float* __restrict__ wq, const float* __restrict__ wk,
            const float* __restrict__ wv, const float* __restrict__ wo,
            u16* __restrict__ o /* wq|wk|wv|wo contiguous, 65536 each */) {
    int i = blockIdx.x * 256 + threadIdx.x;     // 0 .. 262143
    int w = i >> 16, j = i & 65535;
    const float* src = (w == 0) ? wq : (w == 1) ? wk : (w == 2) ? wv : wo;
    o[i] = f2bf(src[j]);
}

// ------------------------------------------------- LayerNorm + Q proj + RoPE(Q)
__global__ __launch_bounds__(128)
void k_q_proj(const float* __restrict__ query, const float* __restrict__ qpos,
              const float* __restrict__ nscale, const float* __restrict__ nbias,
              const u16* __restrict__ wq_bf, u16* __restrict__ q_bf) {
    __shared__ __align__(16) float sF[16 * 256];   // f32 staging (in, then out)
    __shared__ __align__(16) u16   sA[16 * 256];   // bf16 A tile
    __shared__ float sMu[16], sRs[16];

    int tid = threadIdx.x;
    int row0 = blockIdx.x * 16;                    // global query row

    // hint the weight matrix toward the caches while we do LayerNorm
    __builtin_prefetch(&wq_bf[(size_t)(tid * 2) * 256], 0, 3);
    __builtin_prefetch(&wq_bf[(size_t)(tid * 2 + 1) * 256], 0, 3);

    for (int i = tid; i < 16 * 256; i += 128) sF[i] = query[(size_t)row0 * 256 + i];
    __syncthreads();
    if (tid < 16) {
        float mu = 0.f;
        for (int c = 0; c < 256; ++c) mu += sF[tid * 256 + c];
        mu *= (1.0f / 256.0f);
        float var = 0.f;
        for (int c = 0; c < 256; ++c) { float d = sF[tid * 256 + c] - mu; var += d * d; }
        var *= (1.0f / 256.0f);
        sMu[tid] = mu;
        sRs[tid] = rsqrtf(var + 1e-5f);
    }
    __syncthreads();
    for (int i = tid; i < 16 * 256; i += 128) {
        int r = i >> 8, c = i & 255;
        sA[i] = f2bf((sF[i] - sMu[r]) * sRs[r] * nscale[c] + nbias[c]);
    }
    __syncthreads();

    int wave = tid >> 5, lane = tid & 31;
    int arow = lane & 15;
    int kb16 = (lane & 16) ? 16 : 0;
    int rbase = (lane & 16) ? 8 : 0;

    // hoist A fragments: loaded once, reused by all 4 N-tiles
    v16bf aF[8];
#pragma unroll
    for (int kk = 0; kk < 8; ++kk) aF[kk] = a_frag(&sA[arow * 256 + kk * 32], lane);

    for (int nt = wave * 4; nt < wave * 4 + 4; ++nt) {
        const u16* wp = &wq_bf[(size_t)(nt * 16 + arow) * 256 + kb16];
        v8f acc = {};
#pragma unroll
        for (int kk = 0; kk < 8; ++kk)
            acc = wmma_bf16(aF[kk], b_frag(wp + kk * 32), acc);
#pragma unroll
        for (int r = 0; r < 8; ++r) sF[(r + rbase) * 256 + nt * 16 + arow] = acc[r];
    }
    __syncthreads();

    // RoPE(q) in fp32, store bf16 in (b, h, lq, dh) layout
    for (int i = tid; i < 16 * 128; i += 128) {
        int r = i >> 7, pg = i & 127;
        int h = pg >> 4, p = pg & 15;
        int grow = row0 + r;
        float px = qpos[grow * 2 + 0], py = qpos[grow * 2 + 1];
        float inv_freq = __powf(10000.0f, -(float)(p & 7) * 0.125f);
        float ang = ((p < 8) ? px : py) * inv_freq * 6.2831853071795864f;
        float cs = __cosf(ang), sn = __sinf(ang);
        float x1 = sF[r * 256 + h * 32 + 2 * p];
        float x2 = sF[r * 256 + h * 32 + 2 * p + 1];
        int b  = grow >> 9;        // / LQ_
        int lq = grow & 511;
        u16* dst = q_bf + (((size_t)(b * H_ + h) * LQ_ + lq) * DH_);
        dst[2 * p]     = f2bf(x1 * cs - x2 * sn);
        dst[2 * p + 1] = f2bf(x1 * sn + x2 * cs);
    }
}

// ----------------------------------------- K proj + RoPE(K), V proj (transposed)
__global__ __launch_bounds__(128)
void k_kv_proj(const float* __restrict__ value, const float* __restrict__ vpos,
               const u16* __restrict__ wk_bf, const u16* __restrict__ wv_bf,
               u16* __restrict__ k_bf, u16* __restrict__ vT_bf) {
    __shared__ __align__(16) u16   sA[16 * 256];
    __shared__ __align__(16) float sF[16 * 256];

    int tid = threadIdx.x, wg = blockIdx.x;        // B_*340 workgroups
    int b = wg / (S_ / 16), st = wg % (S_ / 16);
    int s0 = st * 16;

    __builtin_prefetch(&wk_bf[(size_t)(tid * 2) * 256], 0, 3);
    __builtin_prefetch(&wv_bf[(size_t)(tid * 2) * 256], 0, 3);

    const float* vrow = value + ((size_t)b * S_ + s0) * 256;
    for (int i = tid; i < 16 * 256; i += 128) sA[i] = f2bf(vrow[i]);
    __syncthreads();

    int wave = tid >> 5, lane = tid & 31;
    int arow = lane & 15;
    int kb16 = (lane & 16) ? 16 : 0;
    int rbase = (lane & 16) ? 8 : 0;

    // hoist A fragments once; reused by V and K projections on all 4 N-tiles
    v16bf aF[8];
#pragma unroll
    for (int kk = 0; kk < 8; ++kk) aF[kk] = a_frag(&sA[arow * 256 + kk * 32], lane);

    for (int nt = wave * 4; nt < wave * 4 + 4; ++nt) {
        int ncol = nt * 16 + arow;
        // ---- V projection (no RoPE), store transposed (b,h,dh,s)
        {
            const u16* wp = &wv_bf[(size_t)ncol * 256 + kb16];
            v8f accv = {};
#pragma unroll
            for (int kk = 0; kk < 8; ++kk)
                accv = wmma_bf16(aF[kk], b_frag(wp + kk * 32), accv);
            int h = ncol >> 5, dh = ncol & 31;
            u16* dst = vT_bf + (((size_t)(b * H_ + h) * DH_ + dh) * S_ + s0 + rbase);
#pragma unroll
            for (int r = 0; r < 8; ++r) dst[r] = f2bf(accv[r]);
        }
        // ---- K projection -> LDS f32 staging for RoPE
        {
            const u16* wp = &wk_bf[(size_t)ncol * 256 + kb16];
            v8f acck = {};
#pragma unroll
            for (int kk = 0; kk < 8; ++kk)
                acck = wmma_bf16(aF[kk], b_frag(wp + kk * 32), acck);
#pragma unroll
            for (int r = 0; r < 8; ++r) sF[(r + rbase) * 256 + nt * 16 + arow] = acck[r];
        }
    }
    __syncthreads();

    // RoPE(k), store bf16 in (b, h, s, dh) layout
    for (int i = tid; i < 16 * 128; i += 128) {
        int r = i >> 7, pg = i & 127;
        int h = pg >> 4, p = pg & 15;
        int s = s0 + r;
        float px = vpos[s * 2 + 0], py = vpos[s * 2 + 1];
        float inv_freq = __powf(10000.0f, -(float)(p & 7) * 0.125f);
        float ang = ((p < 8) ? px : py) * inv_freq * 6.2831853071795864f;
        float cs = __cosf(ang), sn = __sinf(ang);
        float x1 = sF[r * 256 + h * 32 + 2 * p];
        float x2 = sF[r * 256 + h * 32 + 2 * p + 1];
        u16* dst = k_bf + (((size_t)(b * H_ + h) * S_ + s) * DH_);
        dst[2 * p]     = f2bf(x1 * cs - x2 * sn);
        dst[2 * p + 1] = f2bf(x1 * sn + x2 * cs);
    }
}

// ---------------- flash attention: 4 waves per (b,h,q-tile), keys split mod 4
__global__ __launch_bounds__(128)
void k_attn(const u16* __restrict__ q_bf, const u16* __restrict__ k_bf,
            const u16* __restrict__ vT_bf, const unsigned char* __restrict__ pad,
            u16* __restrict__ ao_bf) {
    __shared__ __align__(16) u16   sP[4 * 16 * 32];     // per-wave P tile
    __shared__ __align__(16) float sAcc[4 * 16 * 32];   // per-wave partial acc
    __shared__ float sM[4 * 16], sL[4 * 16];

    int tid  = threadIdx.x;
    int wave = tid >> 5, lane = tid & 31;
    int id = blockIdx.x;              // B_*H_*(LQ_/16)
    int qt = id & 31;
    int bh = id >> 5;
    int b = bh >> 3, h = bh & 7;
    int arow  = lane & 15;
    int kb8   = (lane & 16) ? 8 : 0;
    int kb16  = (lane & 16) ? 16 : 0;
    int rbase = (lane & 16) ? 8 : 0;

    // Q fragment held in registers for the whole loop (A: 16 queries x 32 dh)
    const u16* qp = q_bf + (((size_t)bh * LQ_ + qt * 16 + arow) * DH_);
    Frag qa;
    qa.u[0] = *(const uint4*)(qp + kb16);
    qa.u[1] = *(const uint4*)(qp + kb16 + 8);
    v16bf qfrag = qa.v;

    float mrow[8], lrow[8];
    v8f acc0 = {}, acc1 = {};
#pragma unroll
    for (int r = 0; r < 8; ++r) { mrow[r] = -1e30f; lrow[r] = 0.f; }

    const float sc = 0.17677669529663689f;           // 1/sqrt(DH)
    const u16* kbp = k_bf + (size_t)bh * S_ * DH_;
    const u16* vbp = vT_bf + (size_t)bh * DH_ * S_;
    const unsigned char* padb = pad + (size_t)b * S_;
    u16* myP = &sP[wave * 512];

    // each wave handles key chunks c = wave, wave+4, ... (170 chunks total)
    for (int c = wave; c < S_ / 32; c += 4) {
        int s0 = c * 32;
        // two 16-key score WMMAs: scores = q @ k^T
        v8f s0v = {}, s1v = {};
        {
            const u16* kp0 = kbp + ((size_t)(s0 + arow)) * DH_ + kb16;
            const u16* kp1 = kbp + ((size_t)(s0 + 16 + arow)) * DH_ + kb16;
            Frag f0, f1;
            f0.u[0] = *(const uint4*)kp0; f0.u[1] = *(const uint4*)(kp0 + 8);
            f1.u[0] = *(const uint4*)kp1; f1.u[1] = *(const uint4*)(kp1 + 8);
            s0v = wmma_bf16(qfrag, f0.v, s0v);
            s1v = wmma_bf16(qfrag, f1.v, s1v);
        }
        float bias0 = padb[s0 + arow]      ? -1e9f : 0.f;
        float bias1 = padb[s0 + 16 + arow] ? -1e9f : 0.f;

        float p0[8], p1[8], rm[8], rs[8];
#pragma unroll
        for (int r = 0; r < 8; ++r) {
            p0[r] = s0v[r] * sc + bias0;
            p1[r] = s1v[r] * sc + bias1;
            rm[r] = fmaxf(p0[r], p1[r]);
        }
        // per-row max across the 16 lanes of each half-wave
#pragma unroll
        for (int off = 1; off < 16; off <<= 1)
#pragma unroll
            for (int r = 0; r < 8; ++r)
                rm[r] = fmaxf(rm[r], __shfl_xor(rm[r], off, 16));

        float e0[8], e1[8];
#pragma unroll
        for (int r = 0; r < 8; ++r) {
            float mnew = fmaxf(mrow[r], rm[r]);
            float corr = __expf(mrow[r] - mnew);
            mrow[r] = mnew;
            e0[r] = __expf(p0[r] - mnew);
            e1[r] = __expf(p1[r] - mnew);
            rs[r] = e0[r] + e1[r];
            lrow[r] *= corr;
            acc0[r] *= corr;
            acc1[r] *= corr;
        }
#pragma unroll
        for (int off = 1; off < 16; off <<= 1)
#pragma unroll
            for (int r = 0; r < 8; ++r)
                rs[r] += __shfl_xor(rs[r], off, 16);
#pragma unroll
        for (int r = 0; r < 8; ++r) lrow[r] += rs[r];

        // P (16x32 bf16) to this wave's private LDS slot, reload as A fragment
        // (LDS ops of one wave stay in order -> no barrier needed)
#pragma unroll
        for (int r = 0; r < 8; ++r) {
            myP[(r + rbase) * 32 + arow]      = f2bf(e0[r]);
            myP[(r + rbase) * 32 + 16 + arow] = f2bf(e1[r]);
        }
        Frag pa;
        const u16* pp = &myP[arow * 32];
        pa.u[0] = *(const uint4*)(pp + kb16);
        pa.u[1] = *(const uint4*)(pp + kb16 + 8);

        const u16* vp0 = vbp + ((size_t)(arow)) * S_ + s0 + kb16;       // dh 0..15
        const u16* vp1 = vbp + ((size_t)(16 + arow)) * S_ + s0 + kb16;  // dh 16..31
        Frag f0, f1;
        f0.u[0] = *(const uint4*)vp0; f0.u[1] = *(const uint4*)(vp0 + 8);
        f1.u[0] = *(const uint4*)vp1; f1.u[1] = *(const uint4*)(vp1 + 8);
        acc0 = wmma_bf16(pa.v, f0.v, acc0);
        acc1 = wmma_bf16(pa.v, f1.v, acc1);
    }

    // ---- publish per-wave partials
    if (arow == 0) {
#pragma unroll
        for (int r = 0; r < 8; ++r) {
            sM[wave * 16 + r + rbase] = mrow[r];
            sL[wave * 16 + r + rbase] = lrow[r];
        }
    }
#pragma unroll
    for (int r = 0; r < 8; ++r) {
        sAcc[wave * 512 + (r + rbase) * 32 + arow]      = acc0[r];
        sAcc[wave * 512 + (r + rbase) * 32 + 16 + arow] = acc1[r];
    }
    __syncthreads();

    // ---- merge the 4 partials (flash combine) and store bf16 (b*LQ+q, d)
    for (int i = tid; i < 512; i += 128) {
        int row = i >> 5, dh = i & 31;
        float m0 = sM[row], m1 = sM[16 + row], m2 = sM[32 + row], m3 = sM[48 + row];
        float mt = fmaxf(fmaxf(m0, m1), fmaxf(m2, m3));
        float w0 = __expf(m0 - mt), w1 = __expf(m1 - mt);
        float w2 = __expf(m2 - mt), w3 = __expf(m3 - mt);
        float lt = sL[row] * w0 + sL[16 + row] * w1 + sL[32 + row] * w2 + sL[48 + row] * w3;
        float v  = sAcc[row * 32 + dh] * w0 + sAcc[512 + row * 32 + dh] * w1
                 + sAcc[1024 + row * 32 + dh] * w2 + sAcc[1536 + row * 32 + dh] * w3;
        int q = qt * 16 + row;
        ao_bf[(size_t)(b * LQ_ + q) * D_ + h * DH_ + dh] = f2bf(v / lt);
    }
}

// ---------------------------------------------------- output proj + residual add
__global__ __launch_bounds__(128)
void k_out_proj(const u16* __restrict__ ao_bf, const u16* __restrict__ wo_bf,
                const float* __restrict__ query, float* __restrict__ out) {
    __shared__ __align__(16) u16 sA[16 * 256];
    int tid = threadIdx.x;
    int row0 = blockIdx.x * 16;

    __builtin_prefetch(&wo_bf[(size_t)(tid * 2) * 256], 0, 3);
    __builtin_prefetch(&wo_bf[(size_t)(tid * 2 + 1) * 256], 0, 3);

    const u16* src = ao_bf + (size_t)row0 * 256;
    for (int i = tid * 8; i < 16 * 256; i += 128 * 8)
        *(uint4*)&sA[i] = *(const uint4*)&src[i];
    __syncthreads();

    int wave = tid >> 5, lane = tid & 31;
    int arow = lane & 15;
    int kb16 = (lane & 16) ? 16 : 0;
    int rbase = (lane & 16) ? 8 : 0;

    v16bf aF[8];
#pragma unroll
    for (int kk = 0; kk < 8; ++kk) aF[kk] = a_frag(&sA[arow * 256 + kk * 32], lane);

    for (int nt = wave * 4; nt < wave * 4 + 4; ++nt) {
        const u16* wp = &wo_bf[(size_t)(nt * 16 + arow) * 256 + kb16];
        v8f acc = {};
#pragma unroll
        for (int kk = 0; kk < 8; ++kk)
            acc = wmma_bf16(aF[kk], b_frag(wp + kk * 32), acc);
#pragma unroll
        for (int r = 0; r < 8; ++r) {
            int row = row0 + r + rbase;
            int col = nt * 16 + arow;
            out[(size_t)row * 256 + col] = acc[r] + query[(size_t)row * 256 + col];
        }
    }
}

extern "C" void kernel_launch(void* const* d_in, const int* in_sizes, int n_in,
                              void* d_out, int out_size, void* d_ws, size_t ws_size,
                              hipStream_t stream) {
    (void)in_sizes; (void)n_in; (void)out_size; (void)ws_size;
    const float*         query  = (const float*)d_in[0];
    const float*         qpos   = (const float*)d_in[1];
    /* d_in[2]: query_batch_offsets (int64) == b*LQ uniformly; batch derived from row/LQ */
    const float*         value  = (const float*)d_in[3];
    const unsigned char* pad    = (const unsigned char*)d_in[4];
    const float*         vpos   = (const float*)d_in[5];
    const float*         nscale = (const float*)d_in[6];
    const float*         nbias  = (const float*)d_in[7];
    const float*         wq     = (const float*)d_in[8];
    const float*         wk     = (const float*)d_in[9];
    const float*         wv     = (const float*)d_in[10];
    const float*         wo     = (const float*)d_in[11];
    float*               out    = (float*)d_out;

    char* ws = (char*)d_ws;
    u16* wqkv_bf = (u16*)ws;                 // wq|wk|wv|wo contiguous
    u16* wq_bf = wqkv_bf;
    u16* wk_bf = wqkv_bf + 65536;
    u16* wv_bf = wqkv_bf + 2 * 65536;
    u16* wo_bf = wqkv_bf + 3 * 65536;
    ws += (size_t)4 * 65536 * 2;
    u16* q_bf  = (u16*)ws; ws += (size_t)B_ * H_ * LQ_ * DH_ * 2;
    u16* k_bf  = (u16*)ws; ws += (size_t)B_ * H_ * S_ * DH_ * 2;
    u16* vT_bf = (u16*)ws; ws += (size_t)B_ * H_ * S_ * DH_ * 2;
    u16* ao_bf = (u16*)ws; ws += (size_t)B_ * LQ_ * D_ * 2;

    k_cvt4<<<1024, 256, 0, stream>>>(wq, wk, wv, wo, wqkv_bf);
    k_q_proj<<<(B_ * LQ_) / 16, 128, 0, stream>>>(query, qpos, nscale, nbias, wq_bf, q_bf);
    k_kv_proj<<<B_ * (S_ / 16), 128, 0, stream>>>(value, vpos, wk_bf, wv_bf, k_bf, vT_bf);
    k_attn<<<B_ * H_ * (LQ_ / 16), 128, 0, stream>>>(q_bf, k_bf, vT_bf, pad, ao_bf);
    k_out_proj<<<(B_ * LQ_) / 16, 128, 0, stream>>>(ao_bf, wo_bf, query, out);
}
